// MambaEncoderLayer_91199335563754
// MI455X (gfx1250) — compile-verified, matrix operand order checked by reading, not run on hardware
//
#include <hip/hip_runtime.h>
#include <hip/hip_bf16.h>

// ---------------- types for WMMA ----------------
typedef __bf16 bf16_t;
typedef __bf16 v16bf __attribute__((ext_vector_type(16)));
typedef __bf16 v8bf  __attribute__((ext_vector_type(8)));
typedef float  v8f   __attribute__((ext_vector_type(8)));

// ---------------- problem constants ----------------
constexpr int BSZ   = 2;
constexpr int SEQL  = 2048;
constexpr int DM    = 1024;          // D_MODEL
constexpr int DST   = 16;            // D_STATE
constexpr int DFF   = 4096;
constexpr int DI    = 2048;          // D_INNER
constexpr int RANK  = 64;            // DT_RANK
constexpr int T     = BSZ * SEQL;    // 4096 tokens
constexpr int NXZ   = 2 * DI;        // 4096
constexpr int DBCW  = RANK + 2 * DST; // 96
constexpr float EPSF = 1.1920928955078125e-07f;

// ---------------- CDNA5 async global->LDS support ----------------
#if defined(__gfx1250__) && __has_builtin(__builtin_amdgcn_global_load_async_to_lds_b128)
#define USE_ASYNC_LDS 1
#else
#define USE_ASYNC_LDS 0
#endif

#if USE_ASYNC_LDS
typedef int vint4 __attribute__((ext_vector_type(4)));
typedef __attribute__((address_space(1))) vint4 g_vint4;   // global <4 x i32>
typedef __attribute__((address_space(3))) vint4 l_vint4;   // LDS    <4 x i32>
__device__ __forceinline__ void async_copy_b128(void* lds_dst, const void* g_src) {
    __builtin_amdgcn_global_load_async_to_lds_b128(
        (g_vint4*)g_src, (l_vint4*)lds_dst, 0, 0);
}
__device__ __forceinline__ void wait_async_le4() {
#if __has_builtin(__builtin_amdgcn_s_wait_asynccnt)
    __builtin_amdgcn_s_wait_asynccnt(4);
#else
    asm volatile("s_wait_asynccnt 0x4" ::: "memory");
#endif
}
__device__ __forceinline__ void wait_async_0() {
#if __has_builtin(__builtin_amdgcn_s_wait_asynccnt)
    __builtin_amdgcn_s_wait_asynccnt(0);
#else
    asm volatile("s_wait_asynccnt 0x0" ::: "memory");
#endif
}
#endif

// ================= bf16 tiled WMMA GEMM =================
// C[M,N] = A[M,K] (bf16, row major) * W[N,K]^T (bf16, row major -> B col n == W row n)
// Block tile 128x128, K-step 32, 256 threads = 8 waves, each wave 4x2 16x16 tiles.
// Double-buffered LDS, async DMA staging (global->LDS) pipelined against WMMA.
#define BM 128
#define BN 128
#define BKK 32

__global__ __launch_bounds__(256)
void k_gemm_bf16_wmma(const bf16_t* __restrict__ A, const bf16_t* __restrict__ W,
                      float* __restrict__ C, int M, int N, int K) {
    __shared__ bf16_t As[2][BM][BKK + 8];
    __shared__ bf16_t Bs[2][BN][BKK + 8];

    const int tid   = threadIdx.x;
    const int lane  = tid & 31;
    const int wave  = tid >> 5;
    const int bm    = blockIdx.y * BM;
    const int bn    = blockIdx.x * BN;
    const int wm    = (wave >> 2) * 64;     // 0 / 64
    const int wn    = (wave & 3) * 32;      // 0 / 32 / 64 / 96
    const int lrow  = lane & 15;
    const int lhalf = lane >> 4;            // 0: K 0-7,16-23  1: K 8-15,24-31

    v8f acc[4][2];
    #pragma unroll
    for (int i = 0; i < 4; ++i)
        #pragma unroll
        for (int j = 0; j < 2; ++j)
            #pragma unroll
            for (int e = 0; e < 8; ++e) acc[i][j][e] = 0.0f;

    const int ldr = tid >> 1;            // 0..127
    const int ldc = (tid & 1) * 16;      // 0 / 16
    const int arow = bm + ldr;                           // M==4096, always in range
    const int brow = (bn + ldr < N) ? (bn + ldr) : (N - 1);  // clamp for N=96 tiles
    const bf16_t* aptr = A + (size_t)arow * K + ldc;
    const bf16_t* bptr = W + (size_t)brow * K + ldc;

    const int nk = K / BKK;

    // stage tile kt into LDS buffer buf
    auto stage = [&](int kt, int buf) {
        const int k0 = kt * BKK;
#if USE_ASYNC_LDS
        async_copy_b128(&As[buf][ldr][ldc],     aptr + k0);
        async_copy_b128(&As[buf][ldr][ldc + 8], aptr + k0 + 8);
        async_copy_b128(&Bs[buf][ldr][ldc],     bptr + k0);
        async_copy_b128(&Bs[buf][ldr][ldc + 8], bptr + k0 + 8);
#else
        uint4 a0 = *(const uint4*)(aptr + k0);
        uint4 a1 = *(const uint4*)(aptr + k0 + 8);
        uint4 b0 = *(const uint4*)(bptr + k0);
        uint4 b1 = *(const uint4*)(bptr + k0 + 8);
        *(uint4*)&As[buf][ldr][ldc]     = a0;
        *(uint4*)&As[buf][ldr][ldc + 8] = a1;
        *(uint4*)&Bs[buf][ldr][ldc]     = b0;
        *(uint4*)&Bs[buf][ldr][ldc + 8] = b1;
        if (kt + 1 < nk) {
            __builtin_prefetch(aptr + k0 + BKK, 0, 1);
            __builtin_prefetch(bptr + k0 + BKK, 0, 1);
        }
#endif
    };

    // prologue: stage tile 0 into buffer 0
    stage(0, 0);

    for (int kt = 0; kt < nk; ++kt) {
        const int cur = kt & 1;
        if (kt + 1 < nk) stage(kt + 1, cur ^ 1);
#if USE_ASYNC_LDS
        if (kt + 1 < nk) wait_async_le4();   // in-order: first 4 (tile kt) landed
        else             wait_async_0();
#endif
        __syncthreads();

        // ---- build fragments matching CDNA5 16-bit WMMA lane layout ----
        v16bf afrag[4];
        #pragma unroll
        for (int mi = 0; mi < 4; ++mi) {
            const int r = wm + mi * 16 + lrow;
            v8bf lo = *(const v8bf*)&As[cur][r][lhalf * 8];
            v8bf hi = *(const v8bf*)&As[cur][r][16 + lhalf * 8];
            v16bf a;
            #pragma unroll
            for (int e = 0; e < 8; ++e) { a[e] = lo[e]; a[8 + e] = hi[e]; }
            afrag[mi] = a;
        }
        v16bf bfrag[2];
        #pragma unroll
        for (int nj = 0; nj < 2; ++nj) {
            const int c = wn + nj * 16 + lrow;
            v8bf lo = *(const v8bf*)&Bs[cur][c][lhalf * 8];
            v8bf hi = *(const v8bf*)&Bs[cur][c][16 + lhalf * 8];
            v16bf b;
            #pragma unroll
            for (int e = 0; e < 8; ++e) { b[e] = lo[e]; b[8 + e] = hi[e]; }
            bfrag[nj] = b;
        }

        #pragma unroll
        for (int mi = 0; mi < 4; ++mi)
            #pragma unroll
            for (int nj = 0; nj < 2; ++nj)
                acc[mi][nj] = __builtin_amdgcn_wmma_f32_16x16x32_bf16(
                    false, afrag[mi], false, bfrag[nj],
                    (short)0, acc[mi][nj], false, false);

        __syncthreads();
    }

    // ---- store: VGPR r -> row base + (lane>=16 ? 8 : 0) + r, col = lane&15 ----
    #pragma unroll
    for (int mi = 0; mi < 4; ++mi) {
        #pragma unroll
        for (int nj = 0; nj < 2; ++nj) {
            const int rbase = bm + wm + mi * 16 + lhalf * 8;
            const int col   = bn + wn + nj * 16 + lrow;
            if (col < N) {
                #pragma unroll
                for (int r = 0; r < 8; ++r)
                    C[(size_t)(rbase + r) * N + col] = acc[mi][nj][r];
            }
        }
    }
}

// ================= elementwise / helper kernels =================

__global__ __launch_bounds__(256)
void k_cast_bf16(const float* __restrict__ src, bf16_t* __restrict__ dst, int n) {
    int i = blockIdx.x * 256 + threadIdx.x;
    if (i < n) dst[i] = (bf16_t)src[i];
}

// RMSNorm over rows of width 1024 (+ optional positional embedding), write bf16
__global__ __launch_bounds__(256)
void k_rmsnorm_bf16(const float* __restrict__ x, const float* __restrict__ w,
                    const float* __restrict__ pos, bf16_t* __restrict__ out) {
    __shared__ float red[256];
    const int row = blockIdx.x;
    const int col = threadIdx.x * 4;
    const float* xr = x + (size_t)row * DM;
    float4 xv = *(const float4*)(xr + col);
    float s = xv.x * xv.x + xv.y * xv.y + xv.z * xv.z + xv.w * xv.w;
    red[threadIdx.x] = s;
    __syncthreads();
    for (int st = 128; st > 0; st >>= 1) {
        if (threadIdx.x < st) red[threadIdx.x] += red[threadIdx.x + st];
        __syncthreads();
    }
    const float scale = rsqrtf(red[0] / (float)DM + EPSF);
    const float* xe = &xv.x;
    #pragma unroll
    for (int j = 0; j < 4; ++j) {
        float o = xe[j] * scale * w[col + j];
        if (pos) o += pos[(size_t)row * DM + col + j];
        out[(size_t)row * DM + col + j] = (bf16_t)o;
    }
}

// depthwise causal conv (taps=4) + bias + SiLU on xz[...,:DI]; write f32 + bf16
__global__ __launch_bounds__(256)
void k_conv_silu(const float* __restrict__ xz, const float* __restrict__ cw,
                 const float* __restrict__ cb, float* __restrict__ xcf,
                 bf16_t* __restrict__ xcb) {
    int i = blockIdx.x * 256 + threadIdx.x;      // T*DI
    if (i >= T * DI) return;
    const int d = i % DI;
    const int t = i / DI;
    const int l = t % SEQL;
    float acc = cb[d];
    #pragma unroll
    for (int k = 0; k < 4; ++k) {
        const int ll = l + k - 3;
        if (ll >= 0)
            acc += xz[(size_t)(t + k - 3) * NXZ + d] * cw[d * 4 + k];
    }
    const float sv = acc / (1.0f + expf(-acc));
    xcf[i] = sv;
    xcb[i] = (bf16_t)sv;
}

// slice dbc[:, :RANK] -> bf16 for dt_proj GEMM
__global__ __launch_bounds__(256)
void k_split_dt(const float* __restrict__ dbc, bf16_t* __restrict__ dtb) {
    int i = blockIdx.x * 256 + threadIdx.x;      // T*RANK
    if (i >= T * RANK) return;
    dtb[i] = (bf16_t)dbc[(size_t)(i >> 6) * DBCW + (i & 63)];
}

// delta = softplus(dtproj_out + dt_proj_b), in place
__global__ __launch_bounds__(256)
void k_softplus_bias(float* __restrict__ v, const float* __restrict__ bias) {
    int i = blockIdx.x * 256 + threadIdx.x;      // T*DI
    if (i >= T * DI) return;
    float x = v[i] + bias[i % DI];
    v[i] = (x > 20.0f) ? x : log1pf(expf(x));
}

// selective scan: one thread per (b, d); 16 states in registers
__global__ __launch_bounds__(256)
void k_scan(const float* __restrict__ delta, const float* __restrict__ xcf,
            const float* __restrict__ dbc, const float* __restrict__ A_log,
            float* __restrict__ ys) {
    const int idx = blockIdx.x * 256 + threadIdx.x;  // 0..BSZ*DI-1
    if (idx >= BSZ * DI) return;
    const int b = idx / DI;
    const int d = idx % DI;
    float Arow[DST];
    #pragma unroll
    for (int n = 0; n < DST; ++n) Arow[n] = -expf(A_log[d * DST + n]);
    float h[DST];
    #pragma unroll
    for (int n = 0; n < DST; ++n) h[n] = 0.0f;
    for (int l = 0; l < SEQL; ++l) {
        const size_t t = (size_t)b * SEQL + l;
        const float dv = delta[t * DI + d];
        const float xv = xcf[t * DI + d];
        const float* bc = dbc + t * DBCW;
        float y = 0.0f;
        #pragma unroll
        for (int n = 0; n < DST; ++n) {
            const float dA = __expf(dv * Arow[n]);
            h[n] = dA * h[n] + dv * bc[RANK + n] * xv;
            y += h[n] * bc[RANK + DST + n];
        }
        ys[t * DI + d] = y;
    }
}

// y = (ys + xc*D) * silu(z); z = xz[..., DI:]; write bf16
__global__ __launch_bounds__(256)
void k_ygate(const float* __restrict__ ys, const float* __restrict__ xcf,
             const float* __restrict__ xz, const float* __restrict__ Dp,
             bf16_t* __restrict__ ybf) {
    int i = blockIdx.x * 256 + threadIdx.x;      // T*DI
    if (i >= T * DI) return;
    const int d = i % DI;
    const int t = i / DI;
    const float z  = xz[(size_t)t * NXZ + DI + d];
    const float yv = ys[i] + xcf[i] * Dp[d];
    ybf[i] = (bf16_t)(yv * z / (1.0f + expf(-z)));
}

// x2 = mamba_out + residual
__global__ __launch_bounds__(256)
void k_add_res(const float* __restrict__ a, const float* __restrict__ b,
               float* __restrict__ o, int n) {
    int i = blockIdx.x * 256 + threadIdx.x;
    if (i < n) o[i] = a[i] + b[i];
}

// g = gelu_exact(v + bias) -> bf16
__global__ __launch_bounds__(256)
void k_bias_gelu(const float* __restrict__ v, const float* __restrict__ bias,
                 bf16_t* __restrict__ g) {
    int i = blockIdx.x * 256 + threadIdx.x;      // T*DFF
    if (i >= T * DFF) return;
    const float x = v[i] + bias[i % DFF];
    g[i] = (bf16_t)(0.5f * x * (1.0f + erff(x * 0.70710678118654752f)));
}

// out = ffn2_out + b2 + x2
__global__ __launch_bounds__(256)
void k_final(const float* __restrict__ v, const float* __restrict__ b2,
             const float* __restrict__ x2, float* __restrict__ out) {
    int i = blockIdx.x * 256 + threadIdx.x;      // T*DM
    if (i >= T * DM) return;
    out[i] = v[i] + b2[i % DM] + x2[i];
}

// ================= host-side launch =================
static inline size_t bump(size_t& off, size_t bytes) {
    size_t o = off;
    off += (bytes + 255) & ~(size_t)255;
    return o;
}

extern "C" void kernel_launch(void* const* d_in, const int* in_sizes, int n_in,
                              void* d_out, int out_size, void* d_ws, size_t ws_size,
                              hipStream_t stream) {
    const float* x        = (const float*)d_in[0];
    const float* pos      = (const float*)d_in[1];
    const float* in_proj  = (const float*)d_in[2];   // [NXZ, DM]
    const float* conv_w   = (const float*)d_in[3];   // [DI, 4]
    const float* conv_b   = (const float*)d_in[4];
    const float* x_proj   = (const float*)d_in[5];   // [96, DI]
    const float* dt_proj  = (const float*)d_in[6];   // [DI, RANK]
    const float* dt_b     = (const float*)d_in[7];
    const float* A_log    = (const float*)d_in[8];   // [DI, DST]
    const float* D_param  = (const float*)d_in[9];
    const float* out_proj = (const float*)d_in[10];  // [DM, DI]
    const float* nm_w     = (const float*)d_in[11];
    const float* nf_w     = (const float*)d_in[12];
    const float* w1       = (const float*)d_in[13];  // [DFF, DM]
    const float* b1       = (const float*)d_in[14];
    const float* w2       = (const float*)d_in[15];  // [DM, DFF]
    const float* b2       = (const float*)d_in[16];
    float* out = (float*)d_out;

    char* ws = (char*)d_ws;
    size_t off = 0;
    // bf16 weights
    bf16_t* wbf_in  = (bf16_t*)(ws + bump(off, (size_t)NXZ * DM * 2));
    bf16_t* wbf_xp  = (bf16_t*)(ws + bump(off, (size_t)DBCW * DI * 2));
    bf16_t* wbf_dt  = (bf16_t*)(ws + bump(off, (size_t)DI * RANK * 2));
    bf16_t* wbf_op  = (bf16_t*)(ws + bump(off, (size_t)DM * DI * 2));
    bf16_t* wbf_w1  = (bf16_t*)(ws + bump(off, (size_t)DFF * DM * 2));
    bf16_t* wbf_w2  = (bf16_t*)(ws + bump(off, (size_t)DM * DFF * 2));
    // activations
    bf16_t* h_bf    = (bf16_t*)(ws + bump(off, (size_t)T * DM * 2));
    float*  big0    = (float*)(ws + bump(off, (size_t)T * NXZ * 4)); // xz, then ffn1 out
    float*  xc_f    = (float*)(ws + bump(off, (size_t)T * DI * 4));
    bf16_t* xc_bf   = (bf16_t*)(ws + bump(off, (size_t)T * DI * 2));
    float*  dbc     = (float*)(ws + bump(off, (size_t)T * DBCW * 4));
    bf16_t* dt_bf   = (bf16_t*)(ws + bump(off, (size_t)T * RANK * 2));
    float*  delta   = (float*)(ws + bump(off, (size_t)T * DI * 4));  // dtproj out -> softplus in place
    float*  ysb     = (float*)(ws + bump(off, (size_t)T * DI * 4));
    bf16_t* y_bf    = (bf16_t*)(ws + bump(off, (size_t)T * DI * 2));
    float*  mo      = (float*)(ws + bump(off, (size_t)T * DM * 4));  // out_proj out, then ffn2 out
    float*  x2      = (float*)(ws + bump(off, (size_t)T * DM * 4));
    bf16_t* h2_bf   = (bf16_t*)(ws + bump(off, (size_t)T * DM * 2));
    bf16_t* g_bf    = (bf16_t*)(ws + bump(off, (size_t)T * DFF * 2));
    (void)ws_size; (void)in_sizes; (void)n_in; (void)out_size;

    auto cgrid = [](size_t n) { return dim3((unsigned)((n + 255) / 256)); };

    // 1) weights -> bf16
    k_cast_bf16<<<cgrid((size_t)NXZ * DM), 256, 0, stream>>>(in_proj, wbf_in, NXZ * DM);
    k_cast_bf16<<<cgrid((size_t)DBCW * DI), 256, 0, stream>>>(x_proj, wbf_xp, DBCW * DI);
    k_cast_bf16<<<cgrid((size_t)DI * RANK), 256, 0, stream>>>(dt_proj, wbf_dt, DI * RANK);
    k_cast_bf16<<<cgrid((size_t)DM * DI), 256, 0, stream>>>(out_proj, wbf_op, DM * DI);
    k_cast_bf16<<<cgrid((size_t)DFF * DM), 256, 0, stream>>>(w1, wbf_w1, DFF * DM);
    k_cast_bf16<<<cgrid((size_t)DM * DFF), 256, 0, stream>>>(w2, wbf_w2, DM * DFF);

    // 2) h = rmsnorm(x)*w + pos  (bf16)
    k_rmsnorm_bf16<<<dim3(T), 256, 0, stream>>>(x, nm_w, pos, h_bf);

    // 3) xz = h @ in_proj^T   [T, NXZ]
    k_gemm_bf16_wmma<<<dim3(NXZ / BN, T / BM), 256, 0, stream>>>(h_bf, wbf_in, big0, T, NXZ, DM);

    // 4) conv + SiLU -> xc (f32 + bf16)
    k_conv_silu<<<cgrid((size_t)T * DI), 256, 0, stream>>>(big0, conv_w, conv_b, xc_f, xc_bf);

    // 5) dbc = xc @ x_proj^T  [T, 96]
    k_gemm_bf16_wmma<<<dim3((DBCW + BN - 1) / BN, T / BM), 256, 0, stream>>>(xc_bf, wbf_xp, dbc, T, DBCW, DI);

    // 6) dt slice -> bf16
    k_split_dt<<<cgrid((size_t)T * RANK), 256, 0, stream>>>(dbc, dt_bf);

    // 7) delta_raw = dt @ dt_proj^T  [T, DI]
    k_gemm_bf16_wmma<<<dim3(DI / BN, T / BM), 256, 0, stream>>>(dt_bf, wbf_dt, delta, T, DI, RANK);

    // 8) delta = softplus(delta_raw + dt_b)
    k_softplus_bias<<<cgrid((size_t)T * DI), 256, 0, stream>>>(delta, dt_b);

    // 9) selective scan
    k_scan<<<dim3((BSZ * DI) / 256), 256, 0, stream>>>(delta, xc_f, dbc, A_log, ysb);

    // 10) gate: y = (ys + xc*D) * silu(z)  -> bf16
    k_ygate<<<cgrid((size_t)T * DI), 256, 0, stream>>>(ysb, xc_f, big0, D_param, y_bf);

    // 11) mamba_out = y @ out_proj^T  [T, DM]
    k_gemm_bf16_wmma<<<dim3(DM / BN, T / BM), 256, 0, stream>>>(y_bf, wbf_op, mo, T, DM, DI);

    // 12) x2 = mamba_out + x
    k_add_res<<<cgrid((size_t)T * DM), 256, 0, stream>>>(mo, x, x2, T * DM);

    // 13) h2 = rmsnorm(x2)*w  (bf16)
    k_rmsnorm_bf16<<<dim3(T), 256, 0, stream>>>(x2, nf_w, nullptr, h2_bf);

    // 14) ffn1 = h2 @ w1^T  [T, DFF]  (reuse big0)
    k_gemm_bf16_wmma<<<dim3(DFF / BN, T / BM), 256, 0, stream>>>(h2_bf, wbf_w1, big0, T, DFF, DM);

    // 15) g = gelu(ffn1 + b1) -> bf16
    k_bias_gelu<<<cgrid((size_t)T * DFF), 256, 0, stream>>>(big0, b1, g_bf);

    // 16) ffn2 = g @ w2^T  [T, DM]  (reuse mo)
    k_gemm_bf16_wmma<<<dim3(DM / BN, T / BM), 256, 0, stream>>>(g_bf, wbf_w2, mo, T, DM, DFF);

    // 17) out = ffn2 + b2 + x2
    k_final<<<cgrid((size_t)T * DM), 256, 0, stream>>>(mo, b2, x2, out);
}
